// SparseAttention_47699906789401
// MI455X (gfx1250) — matrix operand order repaired and checked by reading, not compiled, not run on hardware
//
#include <hip/hip_runtime.h>
#include <hip/hip_bf16.h>

typedef __attribute__((ext_vector_type(16))) __bf16 v16bf;
typedef __attribute__((ext_vector_type(8)))  float  v8f;

#define T_SEQ 4096
#define C_DIM 1024
#define H_NUM 16
#define D_DIM 64

// ---------- WMMA helpers ----------
__device__ __forceinline__ v8f wmma_bf16(v16bf a, v16bf b, v8f c) {
    return __builtin_amdgcn_wmma_f32_16x16x32_bf16(
        false, a, false, b, (short)0, c, false, false);
}

// A-fragment (16x32, M x K): lane holds row M = lane&15.
// lane<16: K = {k0+0..7, k0+16..23}; lane>=16: K = {k0+8..15, k0+24..31}.
__device__ __forceinline__ v16bf load_a_frag(const __bf16* row_k0, int lane) {
    int off = (lane & 16) ? 8 : 0;
    union { v16bf v; uint4 u[2]; } f;
    f.u[0] = *(const uint4*)(row_k0 + off);
    f.u[1] = *(const uint4*)(row_k0 + 16 + off);
    return f.v;
}

// B-fragment (32x16, K x N): lane holds col N = lane&15.
// lane<16: K = {k0+0..7, k0+16..23}; lane>=16: K = {k0+8..15, k0+24..31}.
// ptr = &B_colmajor[n][k0] (K contiguous per column).
__device__ __forceinline__ v16bf load_b_frag(const __bf16* col_k0, int lane) {
    int off = (lane & 16) ? 8 : 0;
    union { v16bf v; uint4 u[2]; } f;
    f.u[0] = *(const uint4*)(col_k0 + off);
    f.u[1] = *(const uint4*)(col_k0 + 16 + off);
    return f.v;
}

// ---------- precision conversion kernels ----------
__global__ void cvt_f32_to_bf16(const float* __restrict__ in,
                                __bf16* __restrict__ out, int n) {
    int i = blockIdx.x * 256 + threadIdx.x;
    if (i < n) out[i] = (__bf16)in[i];
}

// W[K][N] f32 -> Wt[N][K] bf16 (so B-fragment K-runs are contiguous)
__global__ void cvt_f32_to_bf16_T(const float* __restrict__ in,
                                  __bf16* __restrict__ out, int K, int N) {
    int i = blockIdx.x * 256 + threadIdx.x;
    if (i < K * N) {
        int k = i / N, n = i % N;
        out[n * K + k] = (__bf16)in[i];
    }
}

// ---------- QKV GEMM: [4096,1024] x [1024,3072] + bias ----------
// Each wave computes a 16x64 strip: one A fragment reused against 4 B tiles
// per k-step (4x arithmetic intensity vs one tile per wave).
// Writes Q (scaled 1/sqrt(D)) as [H][T][D], K as [H][T][D], V^T as [H][D][T].
__global__ __launch_bounds__(256)
void qkv_gemm(const __bf16* __restrict__ xb, const __bf16* __restrict__ WaT,
              const float* __restrict__ bias,
              __bf16* __restrict__ qb, __bf16* __restrict__ kbuf,
              __bf16* __restrict__ vT) {
    const int NSTRIP = 3 * C_DIM / 64;        // 48 strips along N
    int lane = threadIdx.x & 31;
    int wave = threadIdx.x >> 5;
    int task = blockIdx.x * 8 + wave;
    int tn = task % NSTRIP;                   // 64-wide N strip
    int tm = task / NSTRIP;                   // 16-high M tile
    int lanelo = lane & 15;

    const __bf16* arow = xb + (tm * 16 + lanelo) * C_DIM;
    const __bf16* br0 = WaT + (tn * 64 + 0  + lanelo) * C_DIM;
    const __bf16* br1 = WaT + (tn * 64 + 16 + lanelo) * C_DIM;
    const __bf16* br2 = WaT + (tn * 64 + 32 + lanelo) * C_DIM;
    const __bf16* br3 = WaT + (tn * 64 + 48 + lanelo) * C_DIM;

    v8f acc[4] = {{}, {}, {}, {}};
#pragma unroll 2
    for (int k = 0; k < C_DIM; k += 32) {
        __builtin_prefetch(arow + k + 128, 0, 3);
        __builtin_prefetch(br0 + k + 128, 0, 3);
        v16bf a  = load_a_frag(arow + k, lane);
        acc[0] = wmma_bf16(a, load_b_frag(br0 + k, lane), acc[0]);
        acc[1] = wmma_bf16(a, load_b_frag(br1 + k, lane), acc[1]);
        acc[2] = wmma_bf16(a, load_b_frag(br2 + k, lane), acc[2]);
        acc[3] = wmma_bf16(a, load_b_frag(br3 + k, lane), acc[3]);
    }

    int h8 = (lane & 16) ? 8 : 0;
#pragma unroll
    for (int s = 0; s < 4; ++s) {
        int col = tn * 64 + s * 16 + lanelo;
        float bv = bias[col];
#pragma unroll
        for (int r = 0; r < 8; ++r) {
            int row = tm * 16 + r + h8;
            float v = acc[s][r] + bv;
            if (col < C_DIM) {                     // Q, fold in 1/sqrt(64)
                int h = col >> 6, d = col & 63;
                qb[(h * T_SEQ + row) * D_DIM + d] = (__bf16)(v * 0.125f);
            } else if (col < 2 * C_DIM) {          // K
                int c = col - C_DIM, h = c >> 6, d = c & 63;
                kbuf[(h * T_SEQ + row) * D_DIM + d] = (__bf16)v;
            } else {                               // V, transposed [h][d][t]
                int c = col - 2 * C_DIM, h = c >> 6, d = c & 63;
                vT[(h * D_DIM + d) * T_SEQ + row] = (__bf16)v;
            }
        }
    }
}

// ---------- sparse flash attention ----------
// grid = (T/64, H); block = 128 (4 waves); each wave owns 16 queries of one head.
__global__ __launch_bounds__(128)
void sparse_attn(const __bf16* __restrict__ qb, const __bf16* __restrict__ kbuf,
                 const __bf16* __restrict__ vT, __bf16* __restrict__ yb) {
    int lane   = threadIdx.x & 31;
    int wave   = threadIdx.x >> 5;
    int lanelo = lane & 15;
    int h8     = (lane & 16) ? 8 : 0;
    int head   = blockIdx.y;
    int qbase  = blockIdx.x * 64 + wave * 16;

    const __bf16* Qh = qb   + head * T_SEQ * D_DIM;
    const __bf16* Kh = kbuf + head * T_SEQ * D_DIM;
    const __bf16* Vh = vT   + head * D_DIM * T_SEQ;

    __shared__ __bf16 Plds[4][16][32];

    // Q fragments (16 rows x 64 dims = two K-chunks), loaded once.
    v16bf aq0 = load_a_frag(Qh + (qbase + lanelo) * D_DIM + 0,  lane);
    v16bf aq1 = load_a_frag(Qh + (qbase + lanelo) * D_DIM + 32, lane);

    v8f acc0 = {}, acc1 = {}, acc2 = {}, acc3 = {};
    float mrow[8], lrow[8];
#pragma unroll
    for (int r = 0; r < 8; ++r) { mrow[r] = -1e30f; lrow[r] = 0.0f; }

    auto process = [&](int kbase) {
        // S = Q K^T for 32 keys (two 16-key C tiles)
        v8f s0 = {}, s1 = {};
        s0 = wmma_bf16(aq0, load_b_frag(Kh + (kbase + lanelo) * D_DIM + 0,  lane), s0);
        s0 = wmma_bf16(aq1, load_b_frag(Kh + (kbase + lanelo) * D_DIM + 32, lane), s0);
        s1 = wmma_bf16(aq0, load_b_frag(Kh + (kbase + 16 + lanelo) * D_DIM + 0,  lane), s1);
        s1 = wmma_bf16(aq1, load_b_frag(Kh + (kbase + 16 + lanelo) * D_DIM + 32, lane), s1);

        int j0 = kbase + lanelo;
        int j1 = kbase + 16 + lanelo;
        float p0[8], p1[8], mloc[8];
#pragma unroll
        for (int r = 0; r < 8; ++r) {
            int i = qbase + r + h8;
            int d0 = i - j0; d0 = d0 < 0 ? -d0 : d0;
            int d1 = i - j1; d1 = d1 < 0 ? -d1 : d1;
            bool ok0 = (d0 <= 128) || (i < 32) || (j0 < 32);
            bool ok1 = (d1 <= 128) || (i < 32) || (j1 < 32);
            p0[r] = ok0 ? s0[r] : -1e30f;
            p1[r] = ok1 ? s1[r] : -1e30f;
            mloc[r] = fmaxf(p0[r], p1[r]);
        }
        // row max across the 16 lanes of each half-wave
#pragma unroll
        for (int r = 0; r < 8; ++r)
            for (int off = 1; off < 16; off <<= 1)
                mloc[r] = fmaxf(mloc[r], __shfl_xor(mloc[r], off, 32));

        float corr[8], srow[8];
#pragma unroll
        for (int r = 0; r < 8; ++r) {
            float mnew = fmaxf(mrow[r], mloc[r]);
            corr[r] = __expf(mrow[r] - mnew);
            mrow[r] = mnew;
            p0[r] = __expf(p0[r] - mnew);
            p1[r] = __expf(p1[r] - mnew);
            srow[r] = p0[r] + p1[r];
        }
#pragma unroll
        for (int r = 0; r < 8; ++r)
            for (int off = 1; off < 16; off <<= 1)
                srow[r] += __shfl_xor(srow[r], off, 32);
#pragma unroll
        for (int r = 0; r < 8; ++r) {
            lrow[r] = lrow[r] * corr[r] + srow[r];
            acc0[r] *= corr[r]; acc1[r] *= corr[r];
            acc2[r] *= corr[r]; acc3[r] *= corr[r];
        }
        // P (C-layout) -> LDS row-major bf16 -> A-fragment layout
#pragma unroll
        for (int r = 0; r < 8; ++r) {
            Plds[wave][r + h8][lanelo]      = (__bf16)p0[r];
            Plds[wave][r + h8][16 + lanelo] = (__bf16)p1[r];
        }
        v16bf ap = load_a_frag(&Plds[wave][lanelo][0], lane);
        // O += P * V  (V^T layout: K-runs over keys are contiguous)
        acc0 = wmma_bf16(ap, load_b_frag(Vh + (0  + lanelo) * T_SEQ + kbase, lane), acc0);
        acc1 = wmma_bf16(ap, load_b_frag(Vh + (16 + lanelo) * T_SEQ + kbase, lane), acc1);
        acc2 = wmma_bf16(ap, load_b_frag(Vh + (32 + lanelo) * T_SEQ + kbase, lane), acc2);
        acc3 = wmma_bf16(ap, load_b_frag(Vh + (48 + lanelo) * T_SEQ + kbase, lane), acc3);
    };

    if (qbase < 32) {
        // global query rows attend to everything
        for (int kb = 0; kb < T_SEQ; kb += 32) process(kb);
    } else {
        int lo = qbase - 128; if (lo < 0) lo = 0;
        int kstart = lo & ~31;
        int kend = ((qbase + 143) & ~31) + 32;
        if (kend > T_SEQ) kend = T_SEQ;
        if (kstart > 0) process(0);           // global key columns [0,32)
        for (int kb = kstart; kb < kend; kb += 32) process(kb);
    }

    // normalize and write y as bf16 [T][C] for the projection GEMM
#pragma unroll
    for (int r = 0; r < 8; ++r) {
        float inv = 1.0f / lrow[r];
        int row = qbase + r + h8;
        __bf16* yr = yb + row * C_DIM + head * D_DIM;
        yr[0  + lanelo] = (__bf16)(acc0[r] * inv);
        yr[16 + lanelo] = (__bf16)(acc1[r] * inv);
        yr[32 + lanelo] = (__bf16)(acc2[r] * inv);
        yr[48 + lanelo] = (__bf16)(acc3[r] * inv);
    }
}

// ---------- projection GEMM: out = y @ W_proj + b, f32 output ----------
// 16x64 strip per wave, same A-reuse scheme as qkv_gemm.
__global__ __launch_bounds__(256)
void proj_gemm(const __bf16* __restrict__ yb, const __bf16* __restrict__ WpT,
               const float* __restrict__ bias, float* __restrict__ out) {
    const int NSTRIP = C_DIM / 64;            // 16 strips along N
    int lane = threadIdx.x & 31;
    int wave = threadIdx.x >> 5;
    int task = blockIdx.x * 8 + wave;
    int tn = task % NSTRIP;
    int tm = task / NSTRIP;
    int lanelo = lane & 15;

    const __bf16* arow = yb + (tm * 16 + lanelo) * C_DIM;
    const __bf16* br0 = WpT + (tn * 64 + 0  + lanelo) * C_DIM;
    const __bf16* br1 = WpT + (tn * 64 + 16 + lanelo) * C_DIM;
    const __bf16* br2 = WpT + (tn * 64 + 32 + lanelo) * C_DIM;
    const __bf16* br3 = WpT + (tn * 64 + 48 + lanelo) * C_DIM;

    v8f acc[4] = {{}, {}, {}, {}};
#pragma unroll 2
    for (int k = 0; k < C_DIM; k += 32) {
        __builtin_prefetch(arow + k + 128, 0, 3);
        __builtin_prefetch(br0 + k + 128, 0, 3);
        v16bf a = load_a_frag(arow + k, lane);
        acc[0] = wmma_bf16(a, load_b_frag(br0 + k, lane), acc[0]);
        acc[1] = wmma_bf16(a, load_b_frag(br1 + k, lane), acc[1]);
        acc[2] = wmma_bf16(a, load_b_frag(br2 + k, lane), acc[2]);
        acc[3] = wmma_bf16(a, load_b_frag(br3 + k, lane), acc[3]);
    }

    int h8 = (lane & 16) ? 8 : 0;
#pragma unroll
    for (int s = 0; s < 4; ++s) {
        int col = tn * 64 + s * 16 + lanelo;
        float bv = bias[col];
#pragma unroll
        for (int r = 0; r < 8; ++r) {
            int row = tm * 16 + r + h8;
            out[row * C_DIM + col] = acc[s][r] + bv;
        }
    }
}

// ---------- launcher ----------
extern "C" void kernel_launch(void* const* d_in, const int* in_sizes, int n_in,
                              void* d_out, int out_size, void* d_ws, size_t ws_size,
                              hipStream_t stream) {
    const float* x      = (const float*)d_in[0];   // [1,4096,1024]
    const float* W_attn = (const float*)d_in[1];   // [1024,3072]
    const float* b_attn = (const float*)d_in[2];   // [3072]
    const float* W_proj = (const float*)d_in[3];   // [1024,1024]
    const float* b_proj = (const float*)d_in[4];   // [1024]
    float* out = (float*)d_out;

    char* base = (char*)d_ws;
    const size_t MB = 1024ull * 1024ull;
    __bf16* XB  = (__bf16*)(base + 0 * MB);   // x bf16            8 MB
    __bf16* WAT = (__bf16*)(base + 8 * MB);   // W_attn^T bf16     6 MB
    __bf16* WPT = (__bf16*)(base + 14 * MB);  // W_proj^T bf16     2 MB
    __bf16* QB  = (__bf16*)(base + 16 * MB);  // Q [H][T][D]       8 MB
    __bf16* KB  = (__bf16*)(base + 24 * MB);  // K [H][T][D]       8 MB
    __bf16* VT  = (__bf16*)(base + 32 * MB);  // V^T [H][D][T]     8 MB
    __bf16* YB  = (__bf16*)(base + 40 * MB);  // y  [T][C]         8 MB

    const int nX  = T_SEQ * C_DIM;            // 4,194,304
    const int nWa = C_DIM * 3 * C_DIM;        // 3,145,728
    const int nWp = C_DIM * C_DIM;            // 1,048,576

    cvt_f32_to_bf16 <<<(nX  + 255) / 256, 256, 0, stream>>>(x, XB, nX);
    cvt_f32_to_bf16_T<<<(nWa + 255) / 256, 256, 0, stream>>>(W_attn, WAT, C_DIM, 3 * C_DIM);
    cvt_f32_to_bf16_T<<<(nWp + 255) / 256, 256, 0, stream>>>(W_proj, WPT, C_DIM, C_DIM);

    // (4096/16) M-tiles x (3072/64) N-strips = 12288 wave-tasks -> 1536 blocks
    qkv_gemm<<<1536, 256, 0, stream>>>(XB, WAT, b_attn, QB, KB, VT);

    sparse_attn<<<dim3(T_SEQ / 64, H_NUM), 128, 0, stream>>>(QB, KB, VT, YB);

    // (4096/16) x (1024/64) = 4096 wave-tasks -> 512 blocks
    proj_gemm<<<512, 256, 0, stream>>>(YB, WPT, b_proj, out);
}
